// DNEncoder_10007273800197
// MI455X (gfx1250) — matrix operand order repaired and checked by reading, not compiled, not run on hardware
//
#include <hip/hip_runtime.h>
#include <hip/hip_bf16.h>

typedef __attribute__((ext_vector_type(16))) _Float16 v16h;
typedef __attribute__((ext_vector_type(8)))  float    v8f;

#define HEADS 8
#define CWID  32          // per-head channels (both layers)
#define FDIM  256         // HEADS * CWID

// ---- helpers -------------------------------------------------------------

__device__ __forceinline__ void cvt8(const float* __restrict__ p, v16h& d, int o) {
  // p is 16B aligned by construction (all row strides are multiples of 32 floats,
  // all k-offsets are multiples of 8 floats)
  float4 f0 = ((const float4*)p)[0];
  float4 f1 = ((const float4*)p)[1];
  d[o+0]=(_Float16)f0.x; d[o+1]=(_Float16)f0.y; d[o+2]=(_Float16)f0.z; d[o+3]=(_Float16)f0.w;
  d[o+4]=(_Float16)f1.x; d[o+5]=(_Float16)f1.y; d[o+6]=(_Float16)f1.z; d[o+7]=(_Float16)f1.w;
}

__device__ __forceinline__ void atomicMaxF(float* addr, float val) {
  // order-preserving int trick; slots initialized to -inf
  if (val >= 0.0f) atomicMax((int*)addr, __float_as_int(val));
  else             atomicMin((unsigned int*)addr, __float_as_uint(val));
}

// ---- WMMA dual GEMM: Yl = X @ Wl^T + bl ; Yr = X @ Wr^T + br -------------
// X:[N,K] f32, W*:[256,K] f32, Y*:[N,256] f32.  One wave per 16x16 tile.
// A tile (node rows) is loaded/converted once and fed to two WMMAs.

template<int K>
__global__ void gemm_dual(const float* __restrict__ X,
                          const float* __restrict__ Wl, const float* __restrict__ bl,
                          const float* __restrict__ Wr, const float* __restrict__ br,
                          float* __restrict__ Yl, float* __restrict__ Yr, int N) {
  const int lane = threadIdx.x & 31;
  const int wave = threadIdx.x >> 5;
  const int row0 = blockIdx.x << 4;
  const int col0 = ((blockIdx.y * (blockDim.x >> 5)) + wave) << 4;
  const int m    = lane & 15;
  const int hi   = lane >> 4;          // lane half selects K sub-ranges
  const int colB = col0 + m;           // this lane's B column / C column
  const int rowA = min(row0 + m, N - 1);

  v8f accL = {};
  v8f accR = {};
  #pragma unroll
  for (int kb = 0; kb < K; kb += 32) {
    v16h a, bL, bR;
    // A (16x32, MxK): lanes 0-15 hold K {kb..kb+15},
    //                 lanes 16-31 hold K {kb+16..kb+31}
    const float* pA = X + (size_t)rowA * K + kb + hi * 16;
    cvt8(pA,      a, 0);
    cvt8(pA + 8,  a, 8);
    // B (32x16, KxN): lane's column = colB; lanes 0-15 K {kb..kb+15},
    //                 lanes 16-31 K {kb+16..kb+31}; W row-major so contiguous
    const float* pL = Wl + (size_t)colB * K + kb + hi * 16;
    cvt8(pL,     bL, 0);
    cvt8(pL + 8, bL, 8);
    const float* pR = Wr + (size_t)colB * K + kb + hi * 16;
    cvt8(pR,     bR, 0);
    cvt8(pR + 8, bR, 8);
    accL = __builtin_amdgcn_wmma_f32_16x16x32_f16(false, a, false, bL, (short)0, accL, false, false);
    accR = __builtin_amdgcn_wmma_f32_16x16x32_f16(false, a, false, bR, (short)0, accR, false, false);
  }
  const float bLs = bl[colB];
  const float bRs = br[colB];
  #pragma unroll
  for (int v = 0; v < 8; ++v) {
    const int row = row0 + v + (hi << 3);   // C layout: VGPR v -> M = v + 8*hi
    if (row < N) {
      Yl[(size_t)row * FDIM + colB] = accL[v] + bLs;
      Yr[(size_t)row * FDIM + colB] = accR[v] + bRs;
    }
  }
}

// ---- per-layer scratch init ---------------------------------------------

__global__ void init_buffers(float* __restrict__ smax, float* __restrict__ denom,
                             float* __restrict__ agg, int N) {
  const int i = blockIdx.x * blockDim.x + threadIdx.x;
  if (i < N * FDIM) agg[i] = 0.0f;
  if (i < N * HEADS) { smax[i] = -__builtin_inff(); denom[i] = 0.0f; }
}

// ---- edge pass 1: scores + segment max (one wave per edge) ---------------
// Virtual edges [E, E+N) are the appended self-loops; original self-loops
// (src==dst) are masked, exactly matching the reference.

__global__ void edge_score_max(const float* __restrict__ xl, const float* __restrict__ xr,
                               const int* __restrict__ ei, const float* __restrict__ att,
                               float* __restrict__ score, float* __restrict__ smax,
                               int E, int Etot) {
  const int wid  = (int)((blockIdx.x * blockDim.x + threadIdx.x) >> 5);
  const int lane = threadIdx.x & 31;
  if (wid >= Etot) return;
  int src, dst;
  if (wid < E) {
    src = ei[wid]; dst = ei[E + wid];
    if (src == dst) {  // masked original self loop
      if (lane < HEADS) score[(size_t)wid * HEADS + lane] = -__builtin_inff();
      return;
    }
  } else {
    src = dst = wid - E;
  }
  const float* pl = xl + (size_t)src * FDIM;
  const float* pr = xr + (size_t)dst * FDIM;
  float mysc = 0.0f;
  #pragma unroll
  for (int h = 0; h < HEADS; ++h) {
    float v = pl[h * CWID + lane] + pr[h * CWID + lane];   // coalesced 128B
    v = (v > 0.0f) ? v : 0.2f * v;                          // leaky_relu 0.2
    float p = v * att[h * CWID + lane];
    #pragma unroll
    for (int off = 16; off > 0; off >>= 1) p += __shfl_xor(p, off, 32);
    if (lane == h) mysc = p;                                // lane h keeps head h
  }
  if (lane < HEADS) {
    score[(size_t)wid * HEADS + lane] = mysc;
    atomicMaxF(&smax[(size_t)dst * HEADS + lane], mysc);
  }
}

// ---- edge pass 2: exp + unnormalized aggregation (one wave per edge) -----

__global__ void edge_aggregate(const float* __restrict__ xl,
                               const int* __restrict__ ei,
                               const float* __restrict__ score, const float* __restrict__ smax,
                               float* __restrict__ denom, float* __restrict__ agg,
                               int E, int Etot) {
  const int wid  = (int)((blockIdx.x * blockDim.x + threadIdx.x) >> 5);
  const int lane = threadIdx.x & 31;
  if (wid >= Etot) return;
  int src, dst;
  if (wid < E) {
    src = ei[wid]; dst = ei[E + wid];
    if (src == dst) return;                                 // masked
  } else {
    src = dst = wid - E;
  }
  const int h8 = lane & 7;
  const float a = __expf(score[(size_t)wid * HEADS + h8] -
                         smax [(size_t)dst * HEADS + h8]);
  if (lane < HEADS) atomicAdd(&denom[(size_t)dst * HEADS + lane], a);
  const float* pl = xl  + (size_t)src * FDIM;
  float*       pa = agg + (size_t)dst * FDIM;
  #pragma unroll
  for (int h = 0; h < HEADS; ++h) {
    const float ah = __shfl(a, h, 32);
    atomicAdd(&pa[h * CWID + lane], ah * pl[h * CWID + lane]);  // L2-resident f32 atomics
  }
}

// ---- node finalize: normalize, mean heads, bias, optional leaky ----------

__global__ void node_finalize(const float* __restrict__ agg, const float* __restrict__ denom,
                              const float* __restrict__ bias, float* __restrict__ out,
                              int N, int lrelu) {
  const int t    = blockIdx.x * blockDim.x + threadIdx.x;
  const int node = t >> 5;
  const int lane = t & 31;
  if (node >= N) return;
  float acc = 0.0f;
  #pragma unroll
  for (int h = 0; h < HEADS; ++h) {
    const float d = fmaxf(denom[(size_t)node * HEADS + h], 1e-16f);
    acc += agg[(size_t)node * FDIM + h * CWID + lane] / d;
  }
  acc = acc * (1.0f / HEADS) + bias[lane];
  if (lrelu) acc = (acc > 0.0f) ? acc : 0.01f * acc;
  out[(size_t)node * CWID + lane] = acc;
}

// ---- launcher ------------------------------------------------------------

extern "C" void kernel_launch(void* const* d_in, const int* in_sizes, int n_in,
                              void* d_out, int out_size, void* d_ws, size_t ws_size,
                              hipStream_t stream) {
  const float* x     = (const float*)d_in[0];
  const int*   ei    = (const int*)  d_in[1];
  const float* W1l   = (const float*)d_in[2];
  const float* b1l   = (const float*)d_in[3];
  const float* W1r   = (const float*)d_in[4];
  const float* b1r   = (const float*)d_in[5];
  const float* att1  = (const float*)d_in[6];
  const float* bias1 = (const float*)d_in[7];
  const float* W2l   = (const float*)d_in[8];
  const float* b2l   = (const float*)d_in[9];
  const float* W2r   = (const float*)d_in[10];
  const float* b2r   = (const float*)d_in[11];
  const float* att2  = (const float*)d_in[12];
  const float* bias2 = (const float*)d_in[13];

  const int N    = in_sizes[0] / 64;   // IN_C = 64
  const int E    = in_sizes[1] / 2;    // edge_index is [2, E]
  const int Etot = E + N;              // + appended self loops

  // workspace layout (f32): ~190 MB total, reused across layers
  float* ws = (float*)d_ws;
  size_t off = 0;
  float* xl    = ws + off; off += (size_t)N * FDIM;
  float* xr    = ws + off; off += (size_t)N * FDIM;
  float* agg   = ws + off; off += (size_t)N * FDIM;
  float* smax  = ws + off; off += (size_t)N * HEADS;
  float* denom = ws + off; off += (size_t)N * HEADS;
  float* hbuf  = ws + off; off += (size_t)N * CWID;
  float* score = ws + off; off += (size_t)Etot * HEADS;

  const dim3 gemmGrid((N + 15) / 16, FDIM / 64);          // 4 waves/block * 16 cols
  const int  initBlocks = (N * FDIM + 255) / 256;
  const int  edgeBlocks = (Etot + 7) / 8;                 // 8 waves (edges) per block
  const int  nodeBlocks = (N * 32 + 255) / 256;

  // ---------------- layer 1 (K = 64) ----------------
  gemm_dual<64><<<gemmGrid, 128, 0, stream>>>(x, W1l, b1l, W1r, b1r, xl, xr, N);
  init_buffers<<<initBlocks, 256, 0, stream>>>(smax, denom, agg, N);
  edge_score_max<<<edgeBlocks, 256, 0, stream>>>(xl, xr, ei, att1, score, smax, E, Etot);
  edge_aggregate<<<edgeBlocks, 256, 0, stream>>>(xl, ei, score, smax, denom, agg, E, Etot);
  node_finalize<<<nodeBlocks, 256, 0, stream>>>(agg, denom, bias1, hbuf, N, 1);

  // ---------------- layer 2 (K = 32) ----------------
  gemm_dual<32><<<gemmGrid, 128, 0, stream>>>(hbuf, W2l, b2l, W2r, b2r, xl, xr, N);
  init_buffers<<<initBlocks, 256, 0, stream>>>(smax, denom, agg, N);
  edge_score_max<<<edgeBlocks, 256, 0, stream>>>(xl, xr, ei, att2, score, smax, E, Etot);
  edge_aggregate<<<edgeBlocks, 256, 0, stream>>>(xl, ei, score, smax, denom, agg, E, Etot);
  node_finalize<<<nodeBlocks, 256, 0, stream>>>(agg, denom, bias2, (float*)d_out, N, 0);
}
